// MF_71751723647734
// MI455X (gfx1250) — compile-verified, matrix operand order checked by reading, not run on hardware
//
#include <hip/hip_runtime.h>
#include <hip/hip_bf16.h>
#include <math.h>

#define N_NODES 100000
#define N_EDGES 1000000
#define N_GRAPHS 512
#define DH 64
#define L_LAYERS 3
#define MAXD 10
#define NBUCK (MAXD + 1)
#define TILE 16
#define NT_MAX ((N_NODES + TILE - 1) / TILE + NBUCK)   // 6261 tiles max
#define PERM_PAD (N_NODES + NBUCK * TILE)
#define NB_SCAN ((N_NODES + 255) / 256)                // 391 scan blocks

typedef __attribute__((ext_vector_type(16))) __bf16 v16bf;
typedef __attribute__((ext_vector_type(8)))  float  v8f;

union BVec {
    v16bf v;
    uint4 q[2];
    unsigned u[8];
};

__device__ __forceinline__ v8f zero8() {
    v8f z = {0.f, 0.f, 0.f, 0.f, 0.f, 0.f, 0.f, 0.f};
    return z;
}

// Load WMMA A operand (16x32 bf16 slice, kb selects K-block) from LDS tile
// stored row-major [16 rows][64 bf16]. Per ISA 16-bit A layout:
// lane L: m = L&15, half = L>>4; VGPR v holds K = {base(v) + half*8, +1}.
// (Adjacent pairs are contiguous, so the compiler merges these into
// two ds_load_b128 per operand.)
__device__ __forceinline__ v16bf load_A(const __hip_bfloat16* lds, int lane, int kb) {
    int m = lane & 15, half = lane >> 4;
    BVec a;
#pragma unroll
    for (int v = 0; v < 8; ++v) {
        int kl = ((v < 4) ? 0 : 16) + ((v & 3) * 2) + half * 8;
        int k = kb * 32 + kl;
        a.u[v] = *(const unsigned*)(lds + m * DH + k);  // two consecutive bf16
    }
    return a.v;
}

// Load WMMA B operand: pre-packed so each lane reads 32 contiguous bytes.
__device__ __forceinline__ v16bf load_B(const __hip_bfloat16* pk, int lane) {
    BVec b;
    const uint4* p = (const uint4*)(pk + lane * 16);
    b.q[0] = p[0];
    b.q[1] = p[1];
    return b.v;
}

// ---------------- degree + bucketing kernels ----------------

__global__ void mf_deg(const int* __restrict__ ei, int* __restrict__ deg) {
    int e = blockIdx.x * blockDim.x + threadIdx.x;
    if (e >= N_EDGES) return;
    atomicAdd(&deg[ei[N_EDGES + e]], 1);
}

__global__ void mf_count(const int* __restrict__ deg, int* __restrict__ degC,
                         int* __restrict__ cnt) {
    int i = blockIdx.x * blockDim.x + threadIdx.x;
    if (i >= N_NODES) return;
    int d = min(deg[i], MAXD);
    degC[i] = d;
    atomicAdd(&cnt[d], 1);
}

__global__ void mf_offsets(const int* __restrict__ cnt, int* __restrict__ off) {
    if (blockIdx.x == 0 && threadIdx.x == 0) {
        int s = 0;
        for (int d = 0; d < NBUCK; ++d) {
            off[d] = s;
            s += ((cnt[d] + TILE - 1) >> 4) << 4;  // pad each bucket to 16
        }
        off[NBUCK] = s;
    }
}

__global__ void mf_perm(const int* __restrict__ degC, const int* __restrict__ off,
                        int* __restrict__ cur, int* __restrict__ perm) {
    int i = blockIdx.x * blockDim.x + threadIdx.x;
    if (i >= N_NODES) return;
    int d = degC[i];
    int p = atomicAdd(&cur[d], 1);
    perm[off[d] + p] = i;
}

__global__ void mf_tiledeg(const int* __restrict__ cnt, const int* __restrict__ off,
                           int* __restrict__ tileDeg) {
    int t = blockIdx.x * blockDim.x + threadIdx.x;
    if (t >= NT_MAX) return;
    int res = -1;
#pragma unroll
    for (int d = 0; d < NBUCK; ++d) {
        int s = off[d] >> 4;
        int e = s + ((cnt[d] + TILE - 1) >> 4);
        if (t >= s && t < e) res = d;
    }
    tileDeg[t] = res;
}

// ---------------- CSR build: exclusive prefix sum of deg + edge placement ----

__global__ void mf_scan1(const int* __restrict__ deg, int* __restrict__ nodeOff,
                         int* __restrict__ blockSums) {
    __shared__ int tmp[256];
    int t = threadIdx.x;
    int i = blockIdx.x * 256 + t;
    int v = (i < N_NODES) ? deg[i] : 0;
    tmp[t] = v;
    __syncthreads();
    for (int off = 1; off < 256; off <<= 1) {
        int a = (t >= off) ? tmp[t - off] : 0;
        __syncthreads();
        tmp[t] += a;
        __syncthreads();
    }
    if (i < N_NODES) nodeOff[i] = tmp[t] - v;  // local exclusive scan
    if (t == 255) blockSums[blockIdx.x] = tmp[255];
}

__global__ void mf_scan2(int* __restrict__ blockSums) {
    if (blockIdx.x == 0 && threadIdx.x == 0) {
        int s = 0;
        for (int b = 0; b < NB_SCAN; ++b) {
            int x = blockSums[b];
            blockSums[b] = s;
            s += x;
        }
    }
}

__global__ void mf_scan3(int* __restrict__ nodeOff, const int* __restrict__ blockSums) {
    int i = blockIdx.x * 256 + threadIdx.x;
    if (i < N_NODES) nodeOff[i] += blockSums[blockIdx.x];
    if (i == 0) nodeOff[N_NODES] = N_EDGES;
}

__global__ void mf_place(const int* __restrict__ ei, const int* __restrict__ nodeOff,
                         int* __restrict__ cursor, int* __restrict__ csrSrc) {
    int e = blockIdx.x * blockDim.x + threadIdx.x;
    if (e >= N_EDGES) return;
    int dst = ei[N_EDGES + e];
    int p = atomicAdd(&cursor[dst], 1);
    csrSrc[nodeOff[dst] + p] = ei[e];
}

// ---------------- weight repack: [M][64][64] f32 -> [M][kb2][nt4][lane32][16] bf16 ----

__global__ void mf_repack(const float* __restrict__ W, __hip_bfloat16* __restrict__ Pk,
                          int M) {
    int idx = blockIdx.x * blockDim.x + threadIdx.x;
    if (idx >= M * 4096) return;
    int mtx = idx >> 12;
    int rem = idx & 4095;
    int kb = rem >> 11;
    int nt = (rem >> 9) & 3;
    int lane = (rem >> 4) & 31;
    int j = rem & 15;
    int half = lane >> 4, nl = lane & 15;
    int v = j >> 1, jj = j & 1;
    int kl = ((v < 4) ? 0 : 16) + ((v & 3) * 2) + half * 8 + jj;
    int k = kb * 32 + kl;
    int n = nt * 16 + nl;
    Pk[idx] = __float2bfloat16(W[(mtx * 64 + k) * 64 + n]);
}

// ---------------- atomic-free neighbor gather: h[i] = sum_{s in nbr(i)} x[s] ----
// One 32-thread group per node; lane owns a float2 column pair. All neighbor
// rows are L2-resident (x is 25.6MB << 192MB L2) -> coalesced b64 gathers.

__global__ void mf_gather(const float* __restrict__ xin, const int* __restrict__ nodeOff,
                          const int* __restrict__ csrSrc, float* __restrict__ h) {
    int tid = blockIdx.x * blockDim.x + threadIdx.x;
    int node = tid >> 5;
    int c = tid & 31;
    if (node >= N_NODES) return;
    int st = nodeOff[node], en = nodeOff[node + 1];
    float a0 = 0.f, a1 = 0.f;
    for (int e = st; e < en; ++e) {
        int s = csrSrc[e];
        float2 v = ((const float2*)(xin + (size_t)s * DH))[c];
        a0 += v.x;
        a1 += v.y;
    }
    ((float2*)(h + (size_t)node * DH))[c] = make_float2(a0, a1);
}

// ---------------- per-tile WMMA transform ----------------
// One wave per 16-node tile (all nodes share degree bucket d):
//   z = h @ Wl[l,d] + bl[l,d] + x @ Wr[l,d]; xnew = sigmoid(z)
//   pooled[batch] += xnew @ lin_w[l]

__global__ __launch_bounds__(32)
void mf_transform(const float* __restrict__ xin, const float* __restrict__ h,
                  const int* __restrict__ perm, const int* __restrict__ tileDeg,
                  const int* __restrict__ batch,
                  const __hip_bfloat16* __restrict__ WlPk,
                  const __hip_bfloat16* __restrict__ WrPk,
                  const __hip_bfloat16* __restrict__ LinPk,
                  const float* __restrict__ bl,
                  float* __restrict__ xout, float* __restrict__ pooled, int layer) {
    __shared__ __align__(16) __hip_bfloat16 sH[TILE * DH];
    __shared__ __align__(16) __hip_bfloat16 sX[TILE * DH];
    __shared__ __align__(16) __hip_bfloat16 sZ[TILE * DH];
    __shared__ int sNid[TILE];
    __shared__ int sBatch[TILE];

    int t = blockIdx.x;
    int d = tileDeg[t];
    if (d < 0) return;  // wave-uniform: past last tile
    int lane = threadIdx.x;

    const __hip_bfloat16* wl = WlPk + (size_t)(layer * NBUCK + d) * 4096;
    const __hip_bfloat16* wr = WrPk + (size_t)(layer * NBUCK + d) * 4096;
    // Warm the weight blocks into cache while we stage features into LDS.
    __builtin_prefetch(wl + lane * 128, 0, 1);   // 8KB across 32 lanes
    __builtin_prefetch(wr + lane * 128, 0, 1);

    if (lane < TILE) {
        int nid = perm[t * TILE + lane];
        sNid[lane] = nid;
        sBatch[lane] = (nid >= 0) ? batch[nid] : 0;
    }
    __syncthreads();

    // stage 16x64 f32 -> bf16 tiles into LDS (zeros for pad rows)
    for (int i = lane; i < TILE * (DH / 4); i += 32) {
        int row = i / (DH / 4);
        int c4 = i % (DH / 4);
        int nid = sNid[row];
        float4 hv = make_float4(0.f, 0.f, 0.f, 0.f);
        float4 xv = make_float4(0.f, 0.f, 0.f, 0.f);
        if (nid >= 0) {
            hv = ((const float4*)(h + (size_t)nid * DH))[c4];
            xv = ((const float4*)(xin + (size_t)nid * DH))[c4];
        }
        __hip_bfloat16* ph = sH + row * DH + c4 * 4;
        ph[0] = __float2bfloat16(hv.x); ph[1] = __float2bfloat16(hv.y);
        ph[2] = __float2bfloat16(hv.z); ph[3] = __float2bfloat16(hv.w);
        __hip_bfloat16* px = sX + row * DH + c4 * 4;
        px[0] = __float2bfloat16(xv.x); px[1] = __float2bfloat16(xv.y);
        px[2] = __float2bfloat16(xv.z); px[3] = __float2bfloat16(xv.w);
    }
    __syncthreads();

    v8f acc[4];
#pragma unroll
    for (int nt = 0; nt < 4; ++nt) acc[nt] = zero8();

#pragma unroll
    for (int kb = 0; kb < 2; ++kb) {
        v16bf aH = load_A(sH, lane, kb);
        v16bf aX = load_A(sX, lane, kb);
#pragma unroll
        for (int nt = 0; nt < 4; ++nt) {
            v16bf bL = load_B(wl + (kb * 4 + nt) * 512, lane);
            acc[nt] = __builtin_amdgcn_wmma_f32_16x16x32_bf16(
                false, aH, false, bL, (short)0, acc[nt], false, false);
            v16bf bR = load_B(wr + (kb * 4 + nt) * 512, lane);
            acc[nt] = __builtin_amdgcn_wmma_f32_16x16x32_bf16(
                false, aX, false, bR, (short)0, acc[nt], false, false);
        }
    }

    // bias + sigmoid; write xnew to LDS (bf16, as A operand) and global (f32)
    int nl = lane & 15, half = lane >> 4;
#pragma unroll
    for (int nt = 0; nt < 4; ++nt) {
        int n = nt * 16 + nl;
        float bias = bl[(size_t)(layer * NBUCK + d) * DH + n];
#pragma unroll
        for (int r = 0; r < 8; ++r) {
            int m = r + 8 * half;
            float z = acc[nt][r] + bias;
            float s = 1.0f / (1.0f + __expf(-z));
            sZ[m * DH + n] = __float2bfloat16(s);
            int nid = sNid[m];
            if (nid >= 0) xout[(size_t)nid * DH + n] = s;
        }
    }
    __syncthreads();

    // projection by lin_w[l] + pooled scatter
    const __hip_bfloat16* lw = LinPk + (size_t)layer * 4096;
    v8f accp[4];
#pragma unroll
    for (int nt = 0; nt < 4; ++nt) accp[nt] = zero8();

#pragma unroll
    for (int kb = 0; kb < 2; ++kb) {
        v16bf aZ = load_A(sZ, lane, kb);
#pragma unroll
        for (int nt = 0; nt < 4; ++nt) {
            v16bf bP = load_B(lw + (kb * 4 + nt) * 512, lane);
            accp[nt] = __builtin_amdgcn_wmma_f32_16x16x32_bf16(
                false, aZ, false, bP, (short)0, accp[nt], false, false);
        }
    }

#pragma unroll
    for (int nt = 0; nt < 4; ++nt) {
        int n = nt * 16 + nl;
#pragma unroll
        for (int r = 0; r < 8; ++r) {
            int m = r + 8 * half;
            if (sNid[m] >= 0)
                unsafeAtomicAdd(&pooled[(size_t)sBatch[m] * DH + n], accp[nt][r]);
        }
    }
}

// ---------------- final: out = pooled @ mlp_w + mlp_b ----------------

__global__ __launch_bounds__(32)
void mf_final(const float* __restrict__ pooled,
              const __hip_bfloat16* __restrict__ MlpPk,
              const float* __restrict__ mlp_b, float* __restrict__ out) {
    __shared__ __align__(16) __hip_bfloat16 sP[TILE * DH];
    int t = blockIdx.x;
    int lane = threadIdx.x;

    for (int i = lane; i < TILE * (DH / 4); i += 32) {
        int row = i / (DH / 4);
        int c4 = i % (DH / 4);
        float4 v = ((const float4*)(pooled + (size_t)(t * TILE + row) * DH))[c4];
        __hip_bfloat16* p = sP + row * DH + c4 * 4;
        p[0] = __float2bfloat16(v.x); p[1] = __float2bfloat16(v.y);
        p[2] = __float2bfloat16(v.z); p[3] = __float2bfloat16(v.w);
    }
    __syncthreads();

    v8f acc[4];
#pragma unroll
    for (int nt = 0; nt < 4; ++nt) acc[nt] = zero8();

#pragma unroll
    for (int kb = 0; kb < 2; ++kb) {
        v16bf a = load_A(sP, lane, kb);
#pragma unroll
        for (int nt = 0; nt < 4; ++nt) {
            v16bf b = load_B(MlpPk + (kb * 4 + nt) * 512, lane);
            acc[nt] = __builtin_amdgcn_wmma_f32_16x16x32_bf16(
                false, a, false, b, (short)0, acc[nt], false, false);
        }
    }

    int nl = lane & 15, half = lane >> 4;
#pragma unroll
    for (int nt = 0; nt < 4; ++nt) {
        int n = nt * 16 + nl;
        float bias = mlp_b[n];
#pragma unroll
        for (int r = 0; r < 8; ++r) {
            int g = t * TILE + r + 8 * half;
            out[(size_t)g * DH + n] = acc[nt][r] + bias;
        }
    }
}

// ---------------- host launch ----------------

static inline size_t alignup256(size_t x) { return (x + 255) & ~(size_t)255; }

extern "C" void kernel_launch(void* const* d_in, const int* in_sizes, int n_in,
                              void* d_out, int out_size, void* d_ws, size_t ws_size,
                              hipStream_t stream) {
    const float* x    = (const float*)d_in[0];
    const int*   ei   = (const int*)d_in[1];
    const int*   batch= (const int*)d_in[2];
    const float* Wl   = (const float*)d_in[3];
    const float* bl   = (const float*)d_in[4];
    const float* Wr   = (const float*)d_in[5];
    const float* lin  = (const float*)d_in[6];
    const float* mlpw = (const float*)d_in[7];
    const float* mlpb = (const float*)d_in[8];
    float* out = (float*)d_out;

    char* w = (char*)d_ws;
    size_t o = 0;
    auto take = [&](size_t bytes) -> void* {
        void* p = w + o;
        o = alignup256(o + bytes);
        return p;
    };

    int* deg       = (int*)take((size_t)N_NODES * 4);
    int* degC      = (int*)take((size_t)N_NODES * 4);
    int* meta      = (int*)take(64 * 4);  // cnt @0, cur @16, off @32
    int* perm      = (int*)take((size_t)PERM_PAD * 4);
    int* tileDeg   = (int*)take((size_t)NT_MAX * 4);
    int* nodeOff   = (int*)take((size_t)(N_NODES + 1) * 4);
    int* blockSums = (int*)take((size_t)NB_SCAN * 4);
    int* cursor    = (int*)take((size_t)N_NODES * 4);
    int* csrSrc    = (int*)take((size_t)N_EDGES * 4);
    float* h      = (float*)take((size_t)N_NODES * DH * 4);
    float* xA     = (float*)take((size_t)N_NODES * DH * 4);
    float* xB     = (float*)take((size_t)N_NODES * DH * 4);
    float* pooled = (float*)take((size_t)N_GRAPHS * DH * 4);
    __hip_bfloat16* WlPk  = (__hip_bfloat16*)take((size_t)L_LAYERS * NBUCK * 4096 * 2);
    __hip_bfloat16* WrPk  = (__hip_bfloat16*)take((size_t)L_LAYERS * NBUCK * 4096 * 2);
    __hip_bfloat16* LinPk = (__hip_bfloat16*)take((size_t)L_LAYERS * 4096 * 2);
    __hip_bfloat16* MlpPk = (__hip_bfloat16*)take((size_t)4096 * 2);

    int* cnt = meta;
    int* cur = meta + 16;
    int* off = meta + 32;

    hipMemsetAsync(deg, 0, (size_t)N_NODES * 4, stream);
    hipMemsetAsync(cursor, 0, (size_t)N_NODES * 4, stream);
    hipMemsetAsync(meta, 0, 64 * 4, stream);
    hipMemsetAsync(perm, 0xFF, (size_t)PERM_PAD * 4, stream);  // -1 sentinels
    hipMemsetAsync(pooled, 0, (size_t)N_GRAPHS * DH * 4, stream);

    // degree + buckets (one-time per launch)
    mf_deg<<<(N_EDGES + 255) / 256, 256, 0, stream>>>(ei, deg);
    mf_count<<<(N_NODES + 255) / 256, 256, 0, stream>>>(deg, degC, cnt);
    mf_offsets<<<1, 32, 0, stream>>>(cnt, off);
    mf_perm<<<(N_NODES + 255) / 256, 256, 0, stream>>>(degC, off, cur, perm);
    mf_tiledeg<<<(NT_MAX + 255) / 256, 256, 0, stream>>>(cnt, off, tileDeg);

    // CSR build (one-time per launch): prefix sum + edge placement
    mf_scan1<<<NB_SCAN, 256, 0, stream>>>(deg, nodeOff, blockSums);
    mf_scan2<<<1, 32, 0, stream>>>(blockSums);
    mf_scan3<<<NB_SCAN, 256, 0, stream>>>(nodeOff, blockSums);
    mf_place<<<(N_EDGES + 255) / 256, 256, 0, stream>>>(ei, nodeOff, cursor, csrSrc);

    // weight repack to WMMA B-operand layout
    int mW = L_LAYERS * NBUCK;  // 33 matrices
    mf_repack<<<(mW * 4096 + 255) / 256, 256, 0, stream>>>(Wl, WlPk, mW);
    mf_repack<<<(mW * 4096 + 255) / 256, 256, 0, stream>>>(Wr, WrPk, mW);
    mf_repack<<<(L_LAYERS * 4096 + 255) / 256, 256, 0, stream>>>(lin, LinPk, L_LAYERS);
    mf_repack<<<(4096 + 255) / 256, 256, 0, stream>>>(mlpw, MlpPk, 1);

    const float* xin = x;
    float* xo[L_LAYERS] = {xA, xB, xA};
    for (int l = 0; l < L_LAYERS; ++l) {
        mf_gather<<<(N_NODES * 32 + 255) / 256, 256, 0, stream>>>(xin, nodeOff, csrSrc, h);
        mf_transform<<<NT_MAX, 32, 0, stream>>>(xin, h, perm, tileDeg, batch, WlPk,
                                                WrPk, LinPk, bl, xo[l], pooled, l);
        xin = xo[l];
    }

    mf_final<<<N_GRAPHS / TILE, 32, 0, stream>>>(pooled, MlpPk, mlpb, out);
}
